// SelfAttention_46763603919067
// MI455X (gfx1250) — matrix operand order repaired and checked by reading, not compiled, not run on hardware
//
#include <hip/hip_runtime.h>
#include <hip/hip_bf16.h>

// ---------------------------------------------------------------------------
// Self-attention for MI455X (gfx1250): B=4, S=2048, H=1024, wave32 WMMA bf16
// Round 2: async global->LDS staging (ASYNCcnt) + double buffering + 64x64
// wave tiles (16 WMMA per 16 ds_load_b128).
// ---------------------------------------------------------------------------

typedef __attribute__((ext_vector_type(16))) __bf16 v16bf;
typedef __attribute__((ext_vector_type(8)))  float  v8f;

#define BQ   4
#define SEQ  2048
#define HD   1024
#define MTOT (BQ * SEQ)

// ---- bf16 helpers ----------------------------------------------------------
__device__ __forceinline__ unsigned short f32_to_bf16(float f) {
    unsigned int u = __builtin_bit_cast(unsigned int, f);
    u += 0x7FFFu + ((u >> 16) & 1u);          // round-to-nearest-even
    return (unsigned short)(u >> 16);
}
__device__ __forceinline__ unsigned int pack_bf16x2(float a, float b) {
    return (unsigned int)f32_to_bf16(a) | ((unsigned int)f32_to_bf16(b) << 16);
}

union FragU { v16bf v; uint4 q[2]; unsigned short h[16]; };

// ---- WMMA fragment builders (ISA 7.12.2 layouts, LDS tiles stride 32) ------
// A (16x32 MxK): lane m = lane%16; elems 0..7 -> K = hi*8+0..7, 8..15 -> 16+hi*8+0..7
__device__ __forceinline__ v16bf load_a_frag(const unsigned short* As, int m0, int lane) {
    const int lm = lane & 15, hi = lane >> 4;
    const unsigned short* p = As + (m0 + lm) * 32 + hi * 8;
    FragU f;
    f.q[0] = *(const uint4*)(p);
    f.q[1] = *(const uint4*)(p + 16);
    return f.v;
}
// B (32x16 KxN) from N-major tile: lane n = lane%16; elems 0..7 -> K = hi*8+0..7, 8..15 -> 16+hi*8+0..7
__device__ __forceinline__ v16bf load_b_frag(const unsigned short* Bs, int n0, int lane) {
    const int ln = lane & 15, hi = lane >> 4;
    const unsigned short* p = Bs + (n0 + ln) * 32 + hi * 8;
    FragU f;
    f.q[0] = *(const uint4*)(p);
    f.q[1] = *(const uint4*)(p + 16);
    return f.v;
}

__device__ __forceinline__ v8f wmma_bf16(v16bf a, v16bf b, v8f c) {
    return __builtin_amdgcn_wmma_f32_16x16x32_bf16(false, a, false, b, (short)0, c,
                                                   false, false);
}

// ---- CDNA5 async global->LDS (tracked by ASYNCcnt) -------------------------
// Generic->LDS pointer: low 32 bits of the generic address are the LDS offset.
__device__ __forceinline__ unsigned lds_off(const void* p) {
    return (unsigned)(size_t)p;
}
__device__ __forceinline__ void async_copy_b128(unsigned lds_byte_off,
                                                const unsigned short* g) {
    unsigned long long ga = (unsigned long long)(size_t)g;
    asm volatile("global_load_async_to_lds_b128 %0, %1, off"
                 :: "v"(lds_byte_off), "v"(ga)
                 : "memory");
}
__device__ __forceinline__ void wait_async0() {
    asm volatile("s_wait_asynccnt 0x0" ::: "memory");
}

// Stage a ROWS x 32(bf16) tile: ROWS*4 16B units, 256 threads.
template <int ROWS>
__device__ __forceinline__ void async_stage(unsigned lds_base,
                                            const unsigned short* src,
                                            size_t stride_halves, int t) {
#pragma unroll
    for (int i = 0; i < ROWS / 64; ++i) {           // ROWS*4/256 iterations
        const int u = t + i * 256;
        const int row = u >> 2, col = (u & 3) * 8;
        async_copy_b128(lds_base + (unsigned)u * 16u,
                        src + (size_t)row * stride_halves + col);
    }
}

// ---- synchronous fp32 tile -> bf16 LDS (QKV projection only) ---------------
__device__ __forceinline__ void stage_f32_tile_as_bf16(unsigned short* dst,
                                                       const float* src,
                                                       size_t src_stride, int t) {
    const int row = t >> 1, col = (t & 1) * 16;
    const float4* s = (const float4*)(src + (size_t)row * src_stride + col);
    float4 f0 = s[0], f1 = s[1], f2 = s[2], f3 = s[3];
    uint4 lo, hi;
    lo.x = pack_bf16x2(f0.x, f0.y); lo.y = pack_bf16x2(f0.z, f0.w);
    lo.z = pack_bf16x2(f1.x, f1.y); lo.w = pack_bf16x2(f1.z, f1.w);
    hi.x = pack_bf16x2(f2.x, f2.y); hi.y = pack_bf16x2(f2.z, f2.w);
    hi.z = pack_bf16x2(f3.x, f3.y); hi.w = pack_bf16x2(f3.z, f3.w);
    uint4* d = (uint4*)(dst + row * 32 + col);
    d[0] = lo;
    d[1] = hi;
}

// ===========================================================================
// Kernel 1: fused Q/K/V projection.  z = 0:Q  1:K  2:V(transposed store)
//   C[m,n] = sum_k x[m,k] * W[n,k] + b[n]     (torch Linear: x @ W.T + b)
// Q,K stored bf16 [B*S][H]; V stored transposed bf16 [H][B*S] for the AV GEMM.
// ===========================================================================
__global__ __launch_bounds__(256) void qkv_kernel(
    const float* __restrict__ x,
    const float* __restrict__ Wq, const float* __restrict__ bq,
    const float* __restrict__ Wk, const float* __restrict__ bk,
    const float* __restrict__ Wv, const float* __restrict__ bv,
    unsigned short* __restrict__ Qo, unsigned short* __restrict__ Ko,
    unsigned short* __restrict__ Vt)
{
    __shared__ alignas(16) unsigned short As[128 * 32];
    __shared__ alignas(16) unsigned short Bs[128 * 32];

    const int t = threadIdx.x;
    const int lane = t & 31, wave = t >> 5;
    const int wm = wave >> 2, wn = wave & 3;           // 2x4 wave grid
    const int m_blk = blockIdx.y * 128, n_blk = blockIdx.x * 128;

    const float* W;  const float* bias;
    if (blockIdx.z == 0)      { W = Wq; bias = bq; }
    else if (blockIdx.z == 1) { W = Wk; bias = bk; }
    else                      { W = Wv; bias = bv; }

    v8f acc[4][2];
#pragma unroll
    for (int mi = 0; mi < 4; ++mi)
#pragma unroll
        for (int ni = 0; ni < 2; ++ni)
            acc[mi][ni] = (v8f){0, 0, 0, 0, 0, 0, 0, 0};

    for (int k0 = 0; k0 < HD; k0 += 32) {
        stage_f32_tile_as_bf16(As, x + (size_t)m_blk * HD + k0, HD, t);
        stage_f32_tile_as_bf16(Bs, W + (size_t)n_blk * HD + k0, HD, t);
        __syncthreads();
        v16bf b0 = load_b_frag(Bs, wn * 32 + 0, lane);
        v16bf b1 = load_b_frag(Bs, wn * 32 + 16, lane);
#pragma unroll
        for (int mi = 0; mi < 4; ++mi) {
            v16bf a = load_a_frag(As, wm * 64 + mi * 16, lane);
            acc[mi][0] = wmma_bf16(a, b0, acc[mi][0]);
            acc[mi][1] = wmma_bf16(a, b1, acc[mi][1]);
        }
        __syncthreads();
    }

    const int ln = lane & 15, hi = lane >> 4;
#pragma unroll
    for (int mi = 0; mi < 4; ++mi)
#pragma unroll
        for (int ni = 0; ni < 2; ++ni) {
            const int ng = n_blk + wn * 32 + ni * 16 + ln;
            const float bb = bias[ng];
            const int mb = m_blk + wm * 64 + mi * 16 + hi * 8;
            if (blockIdx.z == 2) {
                // V transposed: Vt[h][b*S+s]; 8 consecutive m per lane -> one b128 store
                uint4 pk;
                pk.x = pack_bf16x2(acc[mi][ni][0] + bb, acc[mi][ni][1] + bb);
                pk.y = pack_bf16x2(acc[mi][ni][2] + bb, acc[mi][ni][3] + bb);
                pk.z = pack_bf16x2(acc[mi][ni][4] + bb, acc[mi][ni][5] + bb);
                pk.w = pack_bf16x2(acc[mi][ni][6] + bb, acc[mi][ni][7] + bb);
                *(uint4*)(Vt + (size_t)ng * MTOT + mb) = pk;
            } else {
                unsigned short* o = (blockIdx.z == 0) ? Qo : Ko;
#pragma unroll
                for (int r = 0; r < 8; ++r)
                    o[(size_t)(mb + r) * HD + ng] = f32_to_bf16(acc[mi][ni][r] + bb);
            }
        }
}

// ===========================================================================
// Async double-buffered bf16 GEMM core: block 256x128, 8 waves of 64x64.
// A: [256 x K] row-major (lda), B: [128 x K] row-major N-major (ldb).
// Per K-step: 16 v_wmma per wave, fed by async global->LDS staging.
// ===========================================================================
template <typename EPI>
__device__ __forceinline__ void gemm_256x128_async(
    const unsigned short* __restrict__ Ag, size_t lda,
    const unsigned short* __restrict__ Bg, size_t ldb,
    int kdim, v8f (&acc)[4][4], int t)
{
    __shared__ alignas(16) unsigned short As[2][256 * 32];   // 2 x 16 KiB
    __shared__ alignas(16) unsigned short Bs[2][128 * 32];   // 2 x 8  KiB

    const int lane = t & 31, wave = t >> 5;
    const int wm = wave >> 1, wn = wave & 1;                 // 4x2 wave grid

    const unsigned asb[2] = { lds_off(&As[0][0]), lds_off(&As[1][0]) };
    const unsigned bsb[2] = { lds_off(&Bs[0][0]), lds_off(&Bs[1][0]) };

#pragma unroll
    for (int mi = 0; mi < 4; ++mi)
#pragma unroll
        for (int ni = 0; ni < 4; ++ni)
            acc[mi][ni] = (v8f){0, 0, 0, 0, 0, 0, 0, 0};

    // prologue: stage K-slab 0 into buffer 0
    async_stage<256>(asb[0], Ag, lda, t);
    async_stage<128>(bsb[0], Bg, ldb, t);

    const int nk = kdim / 32;
    for (int kt = 0; kt < nk; ++kt) {
        const int cur = kt & 1;
        wait_async0();
        __syncthreads();                       // buffer `cur` fully in LDS
        if (kt + 1 < nk) {                     // prefetch next slab into other buffer
            const int k0 = (kt + 1) * 32;
            async_stage<256>(asb[cur ^ 1], Ag + k0, lda, t);
            async_stage<128>(bsb[cur ^ 1], Bg + k0, ldb, t);
        }
        v16bf bf[4];
#pragma unroll
        for (int ni = 0; ni < 4; ++ni)
            bf[ni] = load_b_frag(Bs[cur], wn * 64 + ni * 16, lane);
#pragma unroll
        for (int mi = 0; mi < 4; ++mi) {
            v16bf a = load_a_frag(As[cur], wm * 64 + mi * 16, lane);
#pragma unroll
            for (int ni = 0; ni < 4; ++ni)
                acc[mi][ni] = wmma_bf16(a, bf[ni], acc[mi][ni]);
        }
        __syncthreads();                       // done reading `cur` before reuse
    }
    (void)0;
}

// ===========================================================================
// Kernel 2: scores = Q @ K^T per batch, fused mask (-1e9 pre-scale) + 1/sqrt(H)
// ===========================================================================
__global__ __launch_bounds__(256) void scores_kernel(
    const unsigned short* __restrict__ Q, const unsigned short* __restrict__ Km,
    const int* __restrict__ mask, float* __restrict__ sc)
{
    const int t = threadIdx.x;
    const int lane = t & 31, wave = t >> 5;
    const int wm = wave >> 1, wn = wave & 1;
    const int b = blockIdx.z;
    const int m_blk = blockIdx.y * 256, n_blk = blockIdx.x * 128;

    const unsigned short* Qb = Q  + (size_t)b * SEQ * HD + (size_t)m_blk * HD;
    const unsigned short* Kb = Km + (size_t)b * SEQ * HD + (size_t)n_blk * HD;

    v8f acc[4][4];
    gemm_256x128_async<int>(Qb, HD, Kb, HD, HD, acc, t);

    const int ln = lane & 15, hi = lane >> 4;
    const float inv_sqrt_h = 0.03125f;  // 1/sqrt(1024)
#pragma unroll
    for (int mi = 0; mi < 4; ++mi)
#pragma unroll
        for (int ni = 0; ni < 4; ++ni) {
            const int kg = n_blk + wn * 64 + ni * 16 + ln;
            const int mb = m_blk + wm * 64 + mi * 16 + hi * 8;
#pragma unroll
            for (int r = 0; r < 8; ++r) {
                const size_t off = ((size_t)b * SEQ + (mb + r)) * SEQ + kg;
                float v = acc[mi][ni][r];
                v = (mask[off] == 0) ? -1.0e9f : v;   // mask BEFORE scale (reference)
                sc[off] = v * inv_sqrt_h;
            }
        }
}

// ===========================================================================
// Kernel 3: row softmax (one 256-thread WG per row), writes bf16 attn
// ===========================================================================
__global__ __launch_bounds__(256) void softmax_kernel(
    const float* __restrict__ sc, unsigned short* __restrict__ attn)
{
    __shared__ float red[8];
    const int rowid = blockIdx.x;
    const float* rp = sc + (size_t)rowid * SEQ;
    const int t = threadIdx.x, lane = t & 31, wave = t >> 5;

    float v[8];
    float mx = -3.402823466e38f;
#pragma unroll
    for (int i = 0; i < 8; ++i) {
        v[i] = rp[t + i * 256];
        mx = fmaxf(mx, v[i]);
    }
#pragma unroll
    for (int off = 16; off > 0; off >>= 1) mx = fmaxf(mx, __shfl_xor(mx, off, 32));
    if (lane == 0) red[wave] = mx;
    __syncthreads();
    float m_all = red[0];
#pragma unroll
    for (int j = 1; j < 8; ++j) m_all = fmaxf(m_all, red[j]);
    __syncthreads();

    float sum = 0.0f;
#pragma unroll
    for (int i = 0; i < 8; ++i) {
        v[i] = __expf(v[i] - m_all);
        sum += v[i];
    }
#pragma unroll
    for (int off = 16; off > 0; off >>= 1) sum += __shfl_xor(sum, off, 32);
    if (lane == 0) red[wave] = sum;
    __syncthreads();
    float s_all = 0.0f;
#pragma unroll
    for (int j = 0; j < 8; ++j) s_all += red[j];
    const float inv = 1.0f / s_all;

    unsigned short* op = attn + (size_t)rowid * SEQ;
#pragma unroll
    for (int i = 0; i < 8; ++i)
        op[t + i * 256] = f32_to_bf16(v[i] * inv);
}

// ===========================================================================
// Kernel 4: out = attn @ V  (V pre-transposed -> contiguous-K B operand)
// ===========================================================================
__global__ __launch_bounds__(256) void av_kernel(
    const unsigned short* __restrict__ attn, const unsigned short* __restrict__ Vt,
    float* __restrict__ out)
{
    const int t = threadIdx.x;
    const int lane = t & 31, wave = t >> 5;
    const int wm = wave >> 1, wn = wave & 1;
    const int b = blockIdx.z;
    const int m_blk = blockIdx.y * 256, n_blk = blockIdx.x * 128;

    const unsigned short* Ab = attn + (size_t)b * SEQ * SEQ + (size_t)m_blk * SEQ;
    const unsigned short* Vb = Vt + (size_t)b * SEQ + (size_t)n_blk * MTOT;

    v8f acc[4][4];
    gemm_256x128_async<int>(Ab, SEQ, Vb, MTOT, SEQ, acc, t);

    const int ln = lane & 15, hi = lane >> 4;
#pragma unroll
    for (int mi = 0; mi < 4; ++mi)
#pragma unroll
        for (int ni = 0; ni < 4; ++ni) {
            const int ng = n_blk + wn * 64 + ni * 16 + ln;
            const int mb = m_blk + wm * 64 + mi * 16 + hi * 8;
#pragma unroll
            for (int r = 0; r < 8; ++r)
                out[((size_t)b * SEQ + (mb + r)) * HD + ng] = acc[mi][ni][r];
        }
}

// ===========================================================================
// Launcher
// ===========================================================================
extern "C" void kernel_launch(void* const* d_in, const int* in_sizes, int n_in,
                              void* d_out, int out_size, void* d_ws, size_t ws_size,
                              hipStream_t stream) {
    (void)in_sizes; (void)n_in; (void)out_size; (void)ws_size;

    const float* x    = (const float*)d_in[0];
    const int*   mask = (const int*)d_in[1];
    const float* Wq   = (const float*)d_in[2];
    const float* bq   = (const float*)d_in[3];
    const float* Wk   = (const float*)d_in[4];
    const float* bk   = (const float*)d_in[5];
    const float* Wv   = (const float*)d_in[6];
    const float* bv   = (const float*)d_in[7];
    float* out = (float*)d_out;

    char* ws = (char*)d_ws;
    const size_t SZ_QKV = (size_t)BQ * SEQ * HD * 2;       // 16 MiB each (bf16)
    const size_t SZ_SC  = (size_t)BQ * SEQ * SEQ * 4;      // 64 MiB (fp32 scores)
    unsigned short* Qw = (unsigned short*)(ws);
    unsigned short* Kw = (unsigned short*)(ws + SZ_QKV);
    unsigned short* Vt = (unsigned short*)(ws + 2 * SZ_QKV);
    float*          Sc = (float*)(ws + 3 * SZ_QKV);
    unsigned short* At = (unsigned short*)(ws + 3 * SZ_QKV + SZ_SC);

    qkv_kernel<<<dim3(HD / 128, MTOT / 128, 3), 256, 0, stream>>>(
        x, Wq, bq, Wk, bk, Wv, bv, Qw, Kw, Vt);
    scores_kernel<<<dim3(SEQ / 128, SEQ / 256, BQ), 256, 0, stream>>>(Qw, Kw, mask, Sc);
    softmax_kernel<<<dim3(BQ * SEQ), 256, 0, stream>>>(Sc, At);
    av_kernel<<<dim3(HD / 128, SEQ / 256, BQ), 256, 0, stream>>>(At, Vt, out);
}
